// WanAnimateFaceBlockCrossAttention_14310831030317
// MI455X (gfx1250) — matrix operand order repaired and checked by reading, not compiled, not run on hardware
//
#include <hip/hip_runtime.h>

// ---------------------------------------------------------------------------
// CDNA5 (gfx1250, wave32) implementation of WanAnimate face-block cross attn.
// bf16 WMMA (v_wmma_f32_16x16x32_bf16) for all GEMMs + flash attention.
// Async global->LDS DMA (ASYNCcnt) with double-buffered GEMM tiles.
// ---------------------------------------------------------------------------

typedef __attribute__((ext_vector_type(16))) __bf16 v16bf;
typedef __attribute__((ext_vector_type(8)))  __bf16 v8bf;
typedef __attribute__((ext_vector_type(8)))  float  v8f;
typedef int v4i __attribute__((vector_size(16)));   // int4, matches builtin proto

#define DIMC   2048
#define HEADS  16
#define DHEAD  128
#define FRAMES 16
#define NKV    512
#define SEQ    16384
#define QPF    1024      // queries per frame
#define EPSLN  1e-6f

// ---- async global->LDS copy (CDNA5 GLOBAL_LOAD_ASYNC_TO_LDS_B128) ----------
#if defined(__has_builtin)
#if __has_builtin(__builtin_amdgcn_global_load_async_to_lds_b128) && \
    __has_builtin(__builtin_amdgcn_s_wait_asynccnt)
#define HAS_ASYNC_LDS 1
#endif
#endif
#ifndef HAS_ASYNC_LDS
#define HAS_ASYNC_LDS 0
#endif

typedef __attribute__((address_space(1))) v4i g1_v4i;
typedef __attribute__((address_space(3))) v4i l3_v4i;

static __device__ inline void async_copy16(const void* g, void* l) {
#if HAS_ASYNC_LDS
  __builtin_amdgcn_global_load_async_to_lds_b128((g1_v4i*)g, (l3_v4i*)l, 0, 0);
#else
  *(v8bf*)l = *(const v8bf*)g;   // fallback: load + ds_store
#endif
}
static __device__ inline void async_wait0() {
#if HAS_ASYNC_LDS
  __builtin_amdgcn_s_wait_asynccnt(0);
#endif
}

static __device__ inline v16bf cat8(v8bf lo, v8bf hi) {
  return __builtin_shufflevector(lo, hi, 0,1,2,3,4,5,6,7,8,9,10,11,12,13,14,15);
}
static __device__ inline v8f wmma_bf16(v16bf a, v16bf b, v8f c) {
  // (neg_a, A, neg_b, B, c_mod, C, reuse_a, reuse_b)
  return __builtin_amdgcn_wmma_f32_16x16x32_bf16(false, a, false, b, (short)0, c,
                                                 false, false);
}
static __device__ inline v8f v8f_zero() {
  v8f z;
#pragma unroll
  for (int i = 0; i < 8; ++i) z[i] = 0.0f;
  return z;
}
static __device__ inline float xor_sum16(float v) {
#pragma unroll
  for (int m = 1; m < 16; m <<= 1) v += __shfl_xor(v, m, 32);
  return v;
}
static __device__ inline float xor_max16(float v) {
#pragma unroll
  for (int m = 1; m < 16; m <<= 1) v = fmaxf(v, __shfl_xor(v, m, 32));
  return v;
}

// ---------------------------------------------------------------------------
// 1) LayerNorm (no scale/bias) + cast to bf16.  One 256-thread block per row.
//    Rows [0,16384) -> hidden_states, rows [16384, 24576) -> encoder states.
// ---------------------------------------------------------------------------
__global__ __launch_bounds__(256) void ln_cast_kernel(
    const float* __restrict__ hs, const float* __restrict__ ehs,
    __bf16* __restrict__ hn, __bf16* __restrict__ en)
{
  const int row = blockIdx.x;
  const float* src;
  __bf16* dst;
  if (row < SEQ) { src = hs  + (size_t)row * DIMC;          dst = hn + (size_t)row * DIMC; }
  else           { src = ehs + (size_t)(row - SEQ) * DIMC;  dst = en + (size_t)(row - SEQ) * DIMC; }

  const int tid = threadIdx.x;
  float x[8], s = 0.0f, ss = 0.0f;
#pragma unroll
  for (int i = 0; i < 8; ++i) {
    x[i] = src[tid + i * 256];
    s  += x[i];
    ss += x[i] * x[i];
  }
#pragma unroll
  for (int m = 1; m < 32; m <<= 1) { s += __shfl_xor(s, m, 32); ss += __shfl_xor(ss, m, 32); }

  __shared__ float ps[8], pss[8];
  const int wave = tid >> 5, lane = tid & 31;
  if (lane == 0) { ps[wave] = s; pss[wave] = ss; }
  __syncthreads();
  s = 0.0f; ss = 0.0f;
#pragma unroll
  for (int i = 0; i < 8; ++i) { s += ps[i]; ss += pss[i]; }

  const float mu   = s * (1.0f / DIMC);
  const float var  = ss * (1.0f / DIMC) - mu * mu;
  const float rstd = rsqrtf(var + EPSLN);
#pragma unroll
  for (int i = 0; i < 8; ++i)
    dst[tid + i * 256] = (__bf16)((x[i] - mu) * rstd);
}

// ---------------------------------------------------------------------------
// 2) Weight transpose + cast:  WT[n][k] = (bf16) W[k][n]   (2048 x 2048 each)
// ---------------------------------------------------------------------------
__global__ __launch_bounds__(256) void wt_cast_kernel(
    const float* __restrict__ Wq, const float* __restrict__ Wk,
    const float* __restrict__ Wv, const float* __restrict__ Wo,
    __bf16* __restrict__ WqT, __bf16* __restrict__ WkT,
    __bf16* __restrict__ WvT, __bf16* __restrict__ WoT)
{
  const float* src; __bf16* dst;
  switch (blockIdx.z) {
    case 0:  src = Wq; dst = WqT; break;
    case 1:  src = Wk; dst = WkT; break;
    case 2:  src = Wv; dst = WvT; break;
    default: src = Wo; dst = WoT; break;
  }
  __shared__ float tile[32][33];
  const int tx = threadIdx.x & 31;
  const int ty = threadIdx.x >> 5;          // 0..7
  const int n0 = blockIdx.x * 32;           // N (output of W)
  const int k0 = blockIdx.y * 32;           // K (input of W)
#pragma unroll
  for (int j = 0; j < 4; ++j) {
    int r = ty + j * 8;
    tile[r][tx] = src[(size_t)(k0 + r) * DIMC + n0 + tx];
  }
  __syncthreads();
#pragma unroll
  for (int j = 0; j < 4; ++j) {
    int r = ty + j * 8;
    dst[(size_t)(n0 + r) * DIMC + k0 + tx] = (__bf16)tile[tx][r];
  }
}

// ---------------------------------------------------------------------------
// 3) 128x128-tile bf16 WMMA GEMM:  C[M][N] = A[M][K] * Bt[N][K]^T + bias
//    8 waves: 4 (row groups of 32) x 2 (col groups of 64); each wave owns
//    2x4 tiles of 16x16.  K-step = 32; ping-pong LDS buffers with async
//    global->LDS DMA overlapping the WMMAs of the current tile.
//    MODE 0 (QK): fused per-head RMSNorm (N-tile == one head) -> bf16 out
//    MODE 1 (V):  bf16 out
//    MODE 2 (O):  f32 out (final result)
// ---------------------------------------------------------------------------
enum { MODE_QK = 0, MODE_BF16 = 1, MODE_F32 = 2 };

template <int MODE>
__global__ __launch_bounds__(256) void gemm128_kernel(
    const __bf16* __restrict__ A, const __bf16* __restrict__ Bt,
    const float* __restrict__ bias, const float* __restrict__ gain,
    void* __restrict__ outv, int M, int N, int K)
{
  __shared__ __align__(16) __bf16 As[2][128 * 40];   // 32 K-halves + 8 pad
  __shared__ __align__(16) __bf16 Bs[2][128 * 40];
  __shared__ float rowSq[256];

  const int tid  = threadIdx.x;
  const int lane = tid & 31, wave = tid >> 5;
  const int wrow = wave & 3, wcol = wave >> 2;
  const int h    = lane >> 4, ml = lane & 15;
  const int rowBase = blockIdx.y * 128;
  const int colBase = blockIdx.x * 128;

  v8f acc[2][4];
#pragma unroll
  for (int r = 0; r < 2; ++r)
#pragma unroll
    for (int c = 0; c < 4; ++c) acc[r][c] = v8f_zero();

  // async stage one 128x32 A tile + 128x32 B tile into buffer `buf`
  auto stage = [&](int k0, int buf) {
#pragma unroll
    for (int i = 0; i < 2; ++i) {
      int idx = tid + i * 256;                 // 512 chunks of 8 halves
      int r = idx >> 2, c8 = idx & 3;
      async_copy16(A  + (size_t)(rowBase + r) * K + k0 + c8 * 8,
                   As[buf] + r * 40 + c8 * 8);
      async_copy16(Bt + (size_t)(colBase + r) * K + k0 + c8 * 8,
                   Bs[buf] + r * 40 + c8 * 8);
    }
  };

  const int NK = K >> 5;
  stage(0, 0);
  async_wait0();
  __syncthreads();

  for (int kt = 0; kt < NK; ++kt) {
    const int cur = kt & 1;
    if (kt + 1 < NK) stage((kt + 1) << 5, cur ^ 1);   // DMA next tile under WMMAs

    // ---- fragments (ISA 7.12.2 layouts) + 8 WMMAs ----
    v16bf af[2], bf[4];
#pragma unroll
    for (int r = 0; r < 2; ++r) {
      const __bf16* p = As[cur] + (wrow * 32 + r * 16 + ml) * 40 + 8 * h;
      af[r] = cat8(*(const v8bf*)p, *(const v8bf*)(p + 16));
    }
#pragma unroll
    for (int c = 0; c < 4; ++c) {
      const __bf16* p = Bs[cur] + (wcol * 64 + c * 16 + ml) * 40 + 8 * h;
      bf[c] = cat8(*(const v8bf*)p, *(const v8bf*)(p + 16));
    }
#pragma unroll
    for (int r = 0; r < 2; ++r)
#pragma unroll
      for (int c = 0; c < 4; ++c)
        acc[r][c] = wmma_bf16(af[r], bf[c], acc[r][c]);

    async_wait0();
    __syncthreads();
  }

  // ---------------- epilogue ----------------
  float bcol[4];
#pragma unroll
  for (int c = 0; c < 4; ++c) bcol[c] = bias[colBase + wcol * 64 + c * 16 + ml];

  if (MODE == MODE_F32) {
    float* outf = (float*)outv;
#pragma unroll
    for (int r = 0; r < 2; ++r)
#pragma unroll
      for (int c = 0; c < 4; ++c) {
        int col = colBase + wcol * 64 + c * 16 + ml;
#pragma unroll
        for (int e = 0; e < 8; ++e) {
          int row = rowBase + wrow * 32 + r * 16 + 8 * h + e;
          outf[(size_t)row * N + col] = acc[r][c][e] + bcol[c];
        }
      }
  } else if (MODE == MODE_BF16) {
    __bf16* outb = (__bf16*)outv;
#pragma unroll
    for (int r = 0; r < 2; ++r)
#pragma unroll
      for (int c = 0; c < 4; ++c) {
        int col = colBase + wcol * 64 + c * 16 + ml;
#pragma unroll
        for (int e = 0; e < 8; ++e) {
          int row = rowBase + wrow * 32 + r * 16 + 8 * h + e;
          outb[(size_t)row * N + col] = (__bf16)(acc[r][c][e] + bcol[c]);
        }
      }
  } else {
    // MODE_QK: bias, then per-head RMSNorm over the 128 cols of this block.
#pragma unroll
    for (int r = 0; r < 2; ++r)
#pragma unroll
      for (int c = 0; c < 4; ++c)
#pragma unroll
        for (int e = 0; e < 8; ++e) acc[r][c][e] += bcol[c];

#pragma unroll
    for (int r = 0; r < 2; ++r)
#pragma unroll
      for (int e = 0; e < 8; ++e) {
        float s = 0.0f;
#pragma unroll
        for (int c = 0; c < 4; ++c) { float t = acc[r][c][e]; s += t * t; }
        s = xor_sum16(s);
        if (ml == 0) rowSq[wcol * 128 + wrow * 32 + r * 16 + 8 * h + e] = s;
      }
    __syncthreads();

    __bf16* outb = (__bf16*)outv;
#pragma unroll
    for (int r = 0; r < 2; ++r)
#pragma unroll
      for (int e = 0; e < 8; ++e) {
        int rl = wrow * 32 + r * 16 + 8 * h + e;
        float tot  = rowSq[rl] + rowSq[128 + rl];
        float rstd = rsqrtf(tot * (1.0f / DHEAD) + EPSLN);
        int row = rowBase + rl;
#pragma unroll
        for (int c = 0; c < 4; ++c) {
          int cl = wcol * 64 + c * 16 + ml;           // d within head
          outb[(size_t)row * N + colBase + cl] = (__bf16)(acc[r][c][e] * rstd * gain[cl]);
        }
      }
  }
}

// ---------------------------------------------------------------------------
// 4) Flash cross-attention, one WG per (frame, head, 128-query block).
//    Q fragments pinned in VGPRs; K (async DMA) / V (transposed) / P staged in
//    dynamic LDS; online softmax with 16-lane butterflies + cross-wave LDS.
//    Softmax scale folded into the exponent (scaled-domain running max).
// ---------------------------------------------------------------------------
__global__ __launch_bounds__(256) void attn_kernel(
    const __bf16* __restrict__ Qn, const __bf16* __restrict__ Kn,
    const __bf16* __restrict__ Vb, __bf16* __restrict__ Out)
{
  extern __shared__ char smem[];
  __bf16* Ks  = (__bf16*)smem;              // [key(128)][d(128)+pad8]
  __bf16* Vs  = Ks + 128 * 136;             // [d(128)][key(128)+pad8] (transposed)
  __bf16* Ps  = Vs + 128 * 136;             // [q(128)][key(128)+pad8]
  float*  redA = (float*)(Ps + 128 * 136);  // [2][128] max
  float*  redB = redA + 256;                // [2][128] sum

  const int qb    = blockIdx.x;             // 0..7
  const int head  = blockIdx.y;             // 0..15
  const int frame = blockIdx.z;             // 0..15

  const int tid  = threadIdx.x;
  const int lane = tid & 31, wave = tid >> 5;
  const int wrow = wave & 3, wcol = wave >> 2;
  const int h    = lane >> 4, ml = lane & 15;

  const int qRow0   = frame * QPF + qb * 128;
  const int kRow0   = frame * NKV;
  const int colBase = head * DHEAD;
  const float scale = 0.08838834764831845f;   // 1/sqrt(128)

  // Q A-fragments, loaded once from global (16B chunks at K = 8h, 16+8h).
  v16bf qf[2][4];
#pragma unroll
  for (int r = 0; r < 2; ++r)
#pragma unroll
    for (int kd = 0; kd < 4; ++kd) {
      const __bf16* p =
          Qn + (size_t)(qRow0 + wrow * 32 + r * 16 + ml) * DIMC + colBase + kd * 32 + 8 * h;
      qf[r][kd] = cat8(*(const v8bf*)p, *(const v8bf*)(p + 16));
    }

  v8f oacc[2][4];
  float mrow[2][8], lrow[2][8];
#pragma unroll
  for (int r = 0; r < 2; ++r) {
#pragma unroll
    for (int c = 0; c < 4; ++c) oacc[r][c] = v8f_zero();
#pragma unroll
    for (int e = 0; e < 8; ++e) { mrow[r][e] = -1e30f; lrow[r][e] = 0.0f; }
  }

  for (int kb = 0; kb < 4; ++kb) {
    // ---- stage K (async DMA, as-is) and V (transposed) blocks ----
#pragma unroll
    for (int i = 0; i < 8; ++i) {
      int idx = tid + i * 256;              // 2048 chunks of 8 halves
      int key = idx >> 4, c8 = idx & 15;
      async_copy16(Kn + (size_t)(kRow0 + kb * 128 + key) * DIMC + colBase + c8 * 8,
                   Ks + key * 136 + c8 * 8);
      v8bf vv = *(const v8bf*)(Vb + (size_t)(kRow0 + kb * 128 + key) * DIMC + colBase + c8 * 8);
#pragma unroll
      for (int j = 0; j < 8; ++j) Vs[(c8 * 8 + j) * 136 + key] = vv[j];
    }
    async_wait0();
    __syncthreads();

    // ---- S = Q K^T (128q x 128keys), 32 WMMAs per wave-block ----
    v8f sacc[2][4];
#pragma unroll
    for (int r = 0; r < 2; ++r)
#pragma unroll
      for (int c = 0; c < 4; ++c) sacc[r][c] = v8f_zero();
#pragma unroll
    for (int kd = 0; kd < 4; ++kd) {
      v16bf bk[4];
#pragma unroll
      for (int c = 0; c < 4; ++c) {
        const __bf16* p = Ks + (wcol * 64 + c * 16 + ml) * 136 + kd * 32 + 8 * h;
        bk[c] = cat8(*(const v8bf*)p, *(const v8bf*)(p + 16));
      }
#pragma unroll
      for (int r = 0; r < 2; ++r)
#pragma unroll
        for (int c = 0; c < 4; ++c)
          sacc[r][c] = wmma_bf16(qf[r][kd], bk[c], sacc[r][c]);
    }

    // ---- block row-max (raw domain; scaled later) ----
#pragma unroll
    for (int r = 0; r < 2; ++r)
#pragma unroll
      for (int e = 0; e < 8; ++e) {
        float m = sacc[r][0][e];
#pragma unroll
        for (int c = 1; c < 4; ++c) m = fmaxf(m, sacc[r][c][e]);
        m = xor_max16(m);
        if (ml == 0) redA[wcol * 128 + wrow * 32 + r * 16 + 8 * h + e] = m;
      }
    __syncthreads();

    // ---- online softmax update: p = exp(fma(s, scale, -m_new)) ----
#pragma unroll
    for (int r = 0; r < 2; ++r)
#pragma unroll
      for (int e = 0; e < 8; ++e) {
        int rl = wrow * 32 + r * 16 + 8 * h + e;
        float bmax  = fmaxf(redA[rl], redA[128 + rl]) * scale;
        float mnew  = fmaxf(mrow[r][e], bmax);
        float alpha = __expf(mrow[r][e] - mnew);
        mrow[r][e]  = mnew;
        float pssum = 0.0f;
#pragma unroll
        for (int c = 0; c < 4; ++c) {
          float pv = __expf(fmaf(sacc[r][c][e], scale, -mnew));
          sacc[r][c][e] = pv;
          pssum += pv;
        }
        pssum = xor_sum16(pssum);
        if (ml == 0) redB[wcol * 128 + rl] = pssum;
        lrow[r][e] *= alpha;
#pragma unroll
        for (int c = 0; c < 4; ++c) oacc[r][c][e] *= alpha;
      }
    __syncthreads();
#pragma unroll
    for (int r = 0; r < 2; ++r)
#pragma unroll
      for (int e = 0; e < 8; ++e) {
        int rl = wrow * 32 + r * 16 + 8 * h + e;
        lrow[r][e] += redB[rl] + redB[128 + rl];
      }

    // ---- P (C-layout) -> LDS (row-major) so it can be re-read as A-frags ----
#pragma unroll
    for (int r = 0; r < 2; ++r)
#pragma unroll
      for (int c = 0; c < 4; ++c) {
        int key = wcol * 64 + c * 16 + ml;
#pragma unroll
        for (int e = 0; e < 8; ++e) {
          int q = wrow * 32 + r * 16 + 8 * h + e;
          Ps[q * 136 + key] = (__bf16)sacc[r][c][e];
        }
      }
    __syncthreads();

    // ---- O += P * V  (32 WMMAs per wave-block) ----
#pragma unroll
    for (int kd = 0; kd < 4; ++kd) {
      v16bf pa[2], bv[4];
#pragma unroll
      for (int r = 0; r < 2; ++r) {
        const __bf16* p = Ps + (wrow * 32 + r * 16 + ml) * 136 + kd * 32 + 8 * h;
        pa[r] = cat8(*(const v8bf*)p, *(const v8bf*)(p + 16));
      }
#pragma unroll
      for (int c = 0; c < 4; ++c) {
        const __bf16* p = Vs + (wcol * 64 + c * 16 + ml) * 136 + kd * 32 + 8 * h;
        bv[c] = cat8(*(const v8bf*)p, *(const v8bf*)(p + 16));
      }
#pragma unroll
      for (int r = 0; r < 2; ++r)
#pragma unroll
        for (int c = 0; c < 4; ++c)
          oacc[r][c] = wmma_bf16(pa[r], bv[c], oacc[r][c]);
    }
    __syncthreads();   // before next kb overwrites Ks/Vs/Ps
  }

  // ---- normalize by l and store bf16 attn output [S][INNER] ----
#pragma unroll
  for (int r = 0; r < 2; ++r)
#pragma unroll
    for (int e = 0; e < 8; ++e) {
      int q = wrow * 32 + r * 16 + 8 * h + e;
      float inv = 1.0f / lrow[r][e];
#pragma unroll
      for (int c = 0; c < 4; ++c) {
        int d = wcol * 64 + c * 16 + ml;
        Out[(size_t)(qRow0 + q) * DIMC + colBase + d] = (__bf16)(oacc[r][c][e] * inv);
      }
    }
}

// ---------------------------------------------------------------------------
// Host orchestration (all launches on `stream`; workspace carved from d_ws).
// ---------------------------------------------------------------------------
extern "C" void kernel_launch(void* const* d_in, const int* in_sizes, int n_in,
                              void* d_out, int out_size, void* d_ws, size_t ws_size,
                              hipStream_t stream) {
  (void)in_sizes; (void)n_in; (void)out_size; (void)ws_size;
  const float* hs  = (const float*)d_in[0];
  const float* ehs = (const float*)d_in[1];
  const float* Wq  = (const float*)d_in[2];
  const float* bq  = (const float*)d_in[3];
  const float* Wk  = (const float*)d_in[4];
  const float* bk  = (const float*)d_in[5];
  const float* Wv  = (const float*)d_in[6];
  const float* bv  = (const float*)d_in[7];
  const float* Wo  = (const float*)d_in[8];
  const float* bo  = (const float*)d_in[9];
  const float* gq  = (const float*)d_in[10];
  const float* gk  = (const float*)d_in[11];
  float* out = (float*)d_out;

  char* ws = (char*)d_ws;
  size_t off = 0;
  auto carve = [&](size_t bytes) -> char* {
    char* p = ws + off;
    off += (bytes + 255) & ~(size_t)255;
    return p;
  };
  __bf16* hn   = (__bf16*)carve((size_t)SEQ * DIMC * 2);           // 64 MB
  __bf16* en   = (__bf16*)carve((size_t)FRAMES * NKV * DIMC * 2);  // 32 MB
  __bf16* WqT  = (__bf16*)carve((size_t)DIMC * DIMC * 2);
  __bf16* WkT  = (__bf16*)carve((size_t)DIMC * DIMC * 2);
  __bf16* WvT  = (__bf16*)carve((size_t)DIMC * DIMC * 2);
  __bf16* WoT  = (__bf16*)carve((size_t)DIMC * DIMC * 2);
  __bf16* Qn   = (__bf16*)carve((size_t)SEQ * DIMC * 2);           // 64 MB
  __bf16* Knb  = (__bf16*)carve((size_t)FRAMES * NKV * DIMC * 2);
  __bf16* Vbuf = (__bf16*)carve((size_t)FRAMES * NKV * DIMC * 2);
  __bf16* attnB = hn;  // hn is dead after the Q GEMM; reuse for attn output

  // 1) LayerNorm + cast
  ln_cast_kernel<<<SEQ + FRAMES * NKV, 256, 0, stream>>>(hs, ehs, hn, en);
  // 2) Weight transpose + cast
  wt_cast_kernel<<<dim3(DIMC / 32, DIMC / 32, 4), 256, 0, stream>>>(
      Wq, Wk, Wv, Wo, WqT, WkT, WvT, WoT);
  // 3) Projections (Q/K fuse per-head RMSNorm)
  gemm128_kernel<MODE_QK><<<dim3(DIMC / 128, SEQ / 128), 256, 0, stream>>>(
      hn, WqT, bq, gq, (void*)Qn, SEQ, DIMC, DIMC);
  gemm128_kernel<MODE_QK><<<dim3(DIMC / 128, (FRAMES * NKV) / 128), 256, 0, stream>>>(
      en, WkT, bk, gk, (void*)Knb, FRAMES * NKV, DIMC, DIMC);
  gemm128_kernel<MODE_BF16><<<dim3(DIMC / 128, (FRAMES * NKV) / 128), 256, 0, stream>>>(
      en, WvT, bv, nullptr, (void*)Vbuf, FRAMES * NKV, DIMC, DIMC);
  // 4) Flash cross-attention
  const size_t attn_smem = 3 * 128 * 136 * sizeof(__bf16) + 512 * sizeof(float);
  attn_kernel<<<dim3(QPF / 128, HEADS, FRAMES), 256, attn_smem, stream>>>(
      Qn, Knb, Vbuf, attnB);
  // 5) Output projection -> f32 result
  gemm128_kernel<MODE_F32><<<dim3(DIMC / 128, SEQ / 128), 256, 0, stream>>>(
      attnB, WoT, bo, nullptr, (void*)out, SEQ, DIMC, DIMC);
}
